// SelfAttentionHead_86560770884117
// MI455X (gfx1250) — matrix operand order repaired and checked by reading, not compiled, not run on hardware
//
#include <hip/hip_runtime.h>
#include <hip/hip_bf16.h>
#include <stdint.h>

typedef __attribute__((ext_vector_type(16))) __bf16 v16bf;
typedef __attribute__((ext_vector_type(8)))  float  v8f;
typedef __attribute__((ext_vector_type(4)))  float  vf4;
typedef __attribute__((ext_vector_type(4)))  __bf16 vbf4;
typedef __attribute__((ext_vector_type(2)))  float  vf2;
typedef __attribute__((ext_vector_type(2)))  __bf16 vbf2;
typedef __attribute__((ext_vector_type(4)))  int    v4i;

#define WMMA_BF16(a, b, c) \
  __builtin_amdgcn_wmma_f32_16x16x32_bf16(false, (a), false, (b), (short)0, (c), false, false)

static __device__ __forceinline__ unsigned int pk2bf(float a, float b) {
  vf2 f; f[0] = a; f[1] = b;
  vbf2 r = __builtin_convertvector(f, vbf2);
  return __builtin_bit_cast(unsigned int, r);
}

union FragU {
  v16bf v;
  uint4 q[2];
};

static __device__ __forceinline__ v16bf load_frag(const unsigned short* p0,
                                                  const unsigned short* p1) {
  FragU f;
  f.q[0] = *(const uint4*)p0;
  f.q[1] = *(const uint4*)p1;
  return f.v;
}

// ---- CDNA5 async global->LDS copy (ASYNCcnt-tracked), with sync fallback ----
#if __has_builtin(__builtin_amdgcn_global_load_async_to_lds_b128)
#define HAVE_ASYNC_LDS 1
#endif

static __device__ __forceinline__ void async_copy16(const unsigned short* g,
                                                    unsigned short* l) {
#ifdef HAVE_ASYNC_LDS
  __builtin_amdgcn_global_load_async_to_lds_b128(
      (v4i*)(uintptr_t)g, (v4i*)(uintptr_t)l, 0, 0);
#else
  *(uint4*)l = *(const uint4*)g;
#endif
}

template <int N>
static __device__ __forceinline__ void async_wait() {
#ifdef HAVE_ASYNC_LDS
#if __has_builtin(__builtin_amdgcn_s_wait_asynccnt)
  __builtin_amdgcn_s_wait_asynccnt(N);
#else
  asm volatile("s_wait_asynccnt %0" :: "i"(N) : "memory");
#endif
#endif
}

// ---------------------------------------------------------------------------
// Kernel 0: convert Wq/Wk/Wv (each C*H fp32) to bf16 into ws: [3][C*H].
// The softmax scale (1/sqrt(H)) and log2(e) are folded into Wq here.
// ---------------------------------------------------------------------------
__global__ void attn_cvt_weights(const float* __restrict__ w0,
                                 const float* __restrict__ w1,
                                 const float* __restrict__ w2,
                                 unsigned short* __restrict__ o, int n) {
  int i = (blockIdx.x * blockDim.x + threadIdx.x) * 2;
  if (i < n) {
    const float qs = 0.125f * 1.44269504088896f;  // 1/sqrt(64) * log2(e)
    *(unsigned int*)&o[i]         = pk2bf(w0[i] * qs, w0[i + 1] * qs);
    *(unsigned int*)&o[n + i]     = pk2bf(w1[i], w1[i + 1]);
    *(unsigned int*)&o[2 * n + i] = pk2bf(w2[i], w2[i + 1]);
  }
}

// ---------------------------------------------------------------------------
// Kernel 1: projection.  grid = (BT/128, 3).  block = 256 (8 waves).
// Q, K stored [BT][64]; V stored transposed [B][64][T] so the attention
// kernel can stage it with pure async byte copies.
// ---------------------------------------------------------------------------
__global__ __launch_bounds__(256) void attn_proj(
    const float* __restrict__ x,            // [BT][1024] fp32
    const unsigned short* __restrict__ wb,  // [3][1024][64] bf16
    unsigned short* __restrict__ out,       // Q,K: [BT][64]; V: [B][64][T]
    int BT, int T, int tShift) {
  __shared__ unsigned short xs[128 * 40];   // x tile, row stride 40 bf16 (pad)
  __shared__ unsigned short wt[64 * 32];    // W chunk transposed: [n][k]

  const int t    = threadIdx.x;
  const int lane = t & 31;
  const int wave = t >> 5;
  const int ln   = lane & 15;
  const int hi   = lane >> 4;
  const int row0 = blockIdx.x * 128;

  const unsigned short* W = wb + (size_t)blockIdx.y * 64 * 1024;
  unsigned short* O       = out + (size_t)blockIdx.y * (size_t)BT * 64;

  v8f acc[4] = {};

  for (int kc = 0; kc < 1024; kc += 32) {
    // ---- stage x[row0:row0+128, kc:kc+32] -> bf16 LDS (packed cvt) ----
#pragma unroll
    for (int i = 0; i < 4; ++i) {
      int idx = t + i * 256;                 // float4 index over 128x8
      int r = idx >> 3, c4 = idx & 7;
      vf4 f = *(const vf4*)(x + (size_t)(row0 + r) * 1024 + kc + c4 * 4);
      vbf4 b4 = __builtin_convertvector(f, vbf4);
      *(uint2*)&xs[r * 40 + c4 * 4] = __builtin_bit_cast(uint2, b4);
    }
    // ---- stage W[kc:kc+32, 0:64] transposed -> wt[n][k] ----
    {
      int k = t >> 3, n0 = (t & 7) * 8;
      uint4 wv = *(const uint4*)(W + (size_t)(kc + k) * 64 + n0);
      const unsigned short* s = (const unsigned short*)&wv;
#pragma unroll
      for (int j = 0; j < 8; ++j) wt[(n0 + j) * 32 + k] = s[j];
    }
    __syncthreads();

    const int m0 = wave * 16;
    v16bf a = load_frag(&xs[(m0 + ln) * 40 + 8 * hi],
                        &xs[(m0 + ln) * 40 + 16 + 8 * hi]);
#pragma unroll
    for (int c = 0; c < 4; ++c) {
      v16bf b = load_frag(&wt[(c * 16 + ln) * 32 + 8 * hi],
                          &wt[(c * 16 + ln) * 32 + 16 + 8 * hi]);
      acc[c] = WMMA_BF16(a, b, acc[c]);
    }
    __syncthreads();
  }

  // ---- store result (C layout: M = r + 8*hi, N = ln) ----
  const int m0g = row0 + wave * 16;
  if (blockIdx.y == 2) {
    // V: transposed store, pairs of adjacent tokens pack into one b32.
#pragma unroll
    for (int c = 0; c < 4; ++c)
#pragma unroll
      for (int r = 0; r < 8; r += 2) {
        unsigned int pp = pk2bf(acc[c][r], acc[c][r + 1]);
        int rt = m0g + r + 8 * hi;
        int bb = rt >> tShift;
        int tt = rt & (T - 1);
        *(unsigned int*)&O[((size_t)bb * 64 + c * 16 + ln) * T + tt] = pp;
      }
  } else {
#pragma unroll
    for (int c = 0; c < 4; ++c)
#pragma unroll
      for (int r = 0; r < 8; r += 2) {
        unsigned int pp = pk2bf(acc[c][r], acc[c][r + 1]);
        int row = m0g + r + 8 * hi;
        O[(size_t)row * 64 + c * 16 + ln]       = (unsigned short)pp;
        O[(size_t)(row + 1) * 64 + c * 16 + ln] = (unsigned short)(pp >> 16);
      }
  }
}

// ---------------------------------------------------------------------------
// Kernel 2: fused causal flash attention.  grid = (T/128, B), block = 256.
// Double-buffered async global->LDS pipeline for K and V^T tiles.
// ---------------------------------------------------------------------------
__global__ __launch_bounds__(256) void attn_flash(
    const unsigned short* __restrict__ qb,   // [B*T][64] bf16 (pre-scaled)
    const unsigned short* __restrict__ kb,   // [B*T][64] bf16
    const unsigned short* __restrict__ vbt,  // [B][64][T] bf16 (transposed)
    float* __restrict__ out,                 // [B*T][64] fp32
    int T) {
  __shared__ unsigned short ks[2][32 * 64];   // K tile row-major [key][h]
  __shared__ unsigned short vt[2][64 * 32];   // V tile transposed [h][key]
  __shared__ unsigned short pls[8][16 * 32];  // per-wave P relayout buffer

  const int t    = threadIdx.x;
  const int lane = t & 31;
  const int wave = t >> 5;
  const int ln   = lane & 15;
  const int hi   = lane >> 4;
  const int b    = blockIdx.y;
  const int q0   = blockIdx.x * 128;
  const int qw   = q0 + wave * 16;

  const size_t rowQ = (size_t)(b * T + qw + ln) * 64;
  v16bf qf[2];
#pragma unroll
  for (int h = 0; h < 2; ++h)
    qf[h] = load_frag(&qb[rowQ + 32 * h + 8 * hi],
                      &qb[rowQ + 32 * h + 16 + 8 * hi]);

  v8f acc[4] = {};
  float mrun[8], lrun[8];
#pragma unroll
  for (int r = 0; r < 8; ++r) { mrun[r] = -3.0e38f; lrun[r] = 0.0f; }

  // Per-tile async staging: 2 x b128 DMA per thread (K seg + V^T seg).
  const int vh = t >> 2, vseg = t & 3;
  auto stage = [&](int kt) {
    int bufi = kt & 1;
    async_copy16(kb + (size_t)(b * T + kt * 32) * 64 + t * 8, &ks[bufi][t * 8]);
    async_copy16(vbt + (size_t)(b * 64 + vh) * T + kt * 32 + vseg * 8,
                 &vt[bufi][vh * 32 + vseg * 8]);
  };

  const int ntiles = (q0 + 128) / 32;  // causal: keys < q0+128
  stage(0);
  for (int kt = 0; kt < ntiles; ++kt) {
    const int k0 = kt * 32;
    const int bufi = kt & 1;

    // Issue next tile's DMA, then wait only for the *current* tile's loads
    // (async loads complete in order; allow the 2 just-issued to remain).
    if (kt + 1 < ntiles) {
      stage(kt + 1);
      async_wait<2>();
    } else {
      async_wait<0>();
    }
    __syncthreads();

    // L2 prefetch two tiles ahead.
    if (kt + 2 < ntiles) {
      __builtin_prefetch(kb + (size_t)(b * T + k0 + 64) * 64 + t * 8, 0, 1);
      __builtin_prefetch(vbt + (size_t)(b * 64 + vh) * T + k0 + 64 + vseg * 8, 0, 1);
    }

    if (k0 <= qw + 15) {  // tile not entirely above the causal diagonal
      // ---- S = Q * K^T  (16 queries x 32 keys), already in exp2 domain ----
      v8f S[2] = {};
#pragma unroll
      for (int h = 0; h < 2; ++h) {
#pragma unroll
        for (int kc = 0; kc < 2; ++kc) {
          v16bf bk = load_frag(&ks[bufi][(kc * 16 + ln) * 64 + 32 * h + 8 * hi],
                               &ks[bufi][(kc * 16 + ln) * 64 + 32 * h + 16 + 8 * hi]);
          S[kc] = WMMA_BF16(qf[h], bk, S[kc]);
        }
      }

      // ---- causal mask + online softmax (exp2 domain) ----
      float p[2][8];
      float mnew[8];
#pragma unroll
      for (int r = 0; r < 8; ++r) {
        int m = qw + r + 8 * hi;
        float s0 = S[0][r];
        float s1 = S[1][r];
        if (k0 + ln > m)      s0 = -3.0e38f;
        if (k0 + 16 + ln > m) s1 = -3.0e38f;
        S[0][r] = s0; S[1][r] = s1;
        float mx = fmaxf(s0, s1);
#pragma unroll
        for (int off = 1; off < 16; off <<= 1)
          mx = fmaxf(mx, __shfl_xor(mx, off, 32));
        mnew[r] = fmaxf(mrun[r], mx);
      }
#pragma unroll
      for (int r = 0; r < 8; ++r) {
        float p0 = exp2f(S[0][r] - mnew[r]);
        float p1 = exp2f(S[1][r] - mnew[r]);
        p[0][r] = p0; p[1][r] = p1;
        float rs = p0 + p1;
#pragma unroll
        for (int off = 1; off < 16; off <<= 1)
          rs += __shfl_xor(rs, off, 32);
        float alpha = exp2f(mrun[r] - mnew[r]);
        lrun[r] = lrun[r] * alpha + rs;
        mrun[r] = mnew[r];
#pragma unroll
        for (int c = 0; c < 4; ++c) acc[c][r] *= alpha;
      }

      // ---- relayout P (C layout -> A layout) through per-wave LDS ----
      unsigned short* P = pls[wave];
#pragma unroll
      for (int r = 0; r < 8; ++r) {
        unsigned int pp = pk2bf(p[0][r], p[1][r]);
        P[(r + 8 * hi) * 32 + ln]      = (unsigned short)pp;
        P[(r + 8 * hi) * 32 + 16 + ln] = (unsigned short)(pp >> 16);
      }
      v16bf pf = load_frag(&P[ln * 32 + 8 * hi], &P[ln * 32 + 16 + 8 * hi]);

      // ---- O += P * V ----
#pragma unroll
      for (int c = 0; c < 4; ++c) {
        v16bf bv = load_frag(&vt[bufi][(c * 16 + ln) * 32 + 8 * hi],
                             &vt[bufi][(c * 16 + ln) * 32 + 16 + 8 * hi]);
        acc[c] = WMMA_BF16(pf, bv, acc[c]);
      }
    }
    __syncthreads();
  }

  // ---- epilogue: normalize and store fp32 ----
#pragma unroll
  for (int r = 0; r < 8; ++r) {
    float inv = 1.0f / lrun[r];
    int row = b * T + qw + r + 8 * hi;
#pragma unroll
    for (int c = 0; c < 4; ++c)
      out[(size_t)row * 64 + c * 16 + ln] = acc[c][r] * inv;
  }
}

// ---------------------------------------------------------------------------
extern "C" void kernel_launch(void* const* d_in, const int* in_sizes, int n_in,
                              void* d_out, int out_size, void* d_ws, size_t ws_size,
                              hipStream_t stream) {
  (void)in_sizes; (void)n_in; (void)out_size; (void)ws_size;
  const float* x  = (const float*)d_in[0];
  const float* wq = (const float*)d_in[1];
  const float* wk = (const float*)d_in[2];
  const float* wv = (const float*)d_in[3];

  const int B = 8, T = 2048, C = 1024, H = 64;
  const int BT = B * T;
  const int tShift = 11;  // log2(T)

  unsigned short* wb = (unsigned short*)d_ws;       // [3][C*H]
  unsigned short* qb = wb + (size_t)3 * C * H;      // [BT][H]
  unsigned short* kb = qb + (size_t)BT * H;         // [BT][H]
  unsigned short* vbt = kb + (size_t)BT * H;        // [B][H][T] (transposed)

  attn_cvt_weights<<<dim3((C * H / 2 + 255) / 256), 256, 0, stream>>>(wq, wk, wv, wb, C * H);
  attn_proj<<<dim3(BT / 128, 3), 256, 0, stream>>>(x, wb, qb, BT, T, tShift);
  attn_flash<<<dim3(T / 128, B), 256, 0, stream>>>(qb, kb, vbt, (float*)d_out, T);
}
